// MultiheadAttention_88055419502944
// MI455X (gfx1250) — matrix operand order repaired and checked by reading, not compiled, not run on hardware
//
#include <hip/hip_runtime.h>
#include <math.h>

#define DEVINL __device__ __forceinline__

typedef __bf16 bf16;
typedef __bf16 v16bf __attribute__((ext_vector_type(16)));
typedef float  v8f   __attribute__((ext_vector_type(8)));

union AFrag { v16bf v; uint4 q[2]; };

constexpr int T_  = 2048;
constexpr int B_  = 2;
constexpr int D_  = 1024;
constexpr int H_  = 16;
constexpr int HD_ = 64;
constexpr int M_  = T_ * B_;     // 4096 rows for the projection GEMMs
constexpr int LKV = T_ + 1;      // 2049 valid kv positions (prefix + T)
constexpr int LP  = 2080;        // padded kv length (65 * 32)
constexpr int BN  = 800;         // bottleneck

// ---------------------------------------------------------------------------
// WMMA wrapper: D = A(16x32 bf16) * B(32x16 bf16) + C(16x16 f32)
// ---------------------------------------------------------------------------
DEVINL v8f wmma_bf16(v16bf a, v16bf b, v8f c) {
  return __builtin_amdgcn_wmma_f32_16x16x32_bf16(
      /*neg_a=*/false, a, /*neg_b=*/false, b,
      /*c_mod=*/(short)0, c, /*reuse_a=*/false, /*reuse_b=*/false);
}

// ---------------------------------------------------------------------------
// 16-lane (half-wave) butterfly reductions.  v_permlane16_b32 permutes within
// rows of 16 lanes via a 64-bit nibble select table -> pure VALU, no LDS
// round-trip / dscnt wait (vs ds_bpermute from __shfl_xor).
// ---------------------------------------------------------------------------
#if __has_builtin(__builtin_amdgcn_permlane16)
DEVINL float permlane16_f(float x, unsigned s1, unsigned s2) {
  union { float f; unsigned u; } a, r;
  a.f = x;
  r.u = __builtin_amdgcn_permlane16(a.u, a.u, s1, s2, false, false);
  return r.f;
}
DEVINL float half_reduce_max(float x) {
  x = fmaxf(x, permlane16_f(x, 0x67452301u, 0xEFCDAB89u));  // xor 1
  x = fmaxf(x, permlane16_f(x, 0x54761032u, 0xDCFE98BAu));  // xor 2
  x = fmaxf(x, permlane16_f(x, 0x32107654u, 0xBA98FEDCu));  // xor 4
  x = fmaxf(x, permlane16_f(x, 0xFEDCBA98u, 0x76543210u));  // xor 8
  return x;
}
DEVINL float half_reduce_sum(float x) {
  x += permlane16_f(x, 0x67452301u, 0xEFCDAB89u);
  x += permlane16_f(x, 0x54761032u, 0xDCFE98BAu);
  x += permlane16_f(x, 0x32107654u, 0xBA98FEDCu);
  x += permlane16_f(x, 0xFEDCBA98u, 0x76543210u);
  return x;
}
#else
DEVINL float half_reduce_max(float x) {
  for (int off = 1; off < 16; off <<= 1) x = fmaxf(x, __shfl_xor(x, off, 32));
  return x;
}
DEVINL float half_reduce_sum(float x) {
  for (int off = 1; off < 16; off <<= 1) x += __shfl_xor(x, off, 32);
  return x;
}
#endif

// ---------------------------------------------------------------------------
// Fragment loaders.
// A-frag 16x32 bf16: lane -> row (lane&15); two 8-elem runs at k = (lane>>4)*8
// and +16 relative to kb.  B-frag 32x16 bf16: lane -> col (lane&15); same
// two 8-elem runs at k = (lane>>4)*8 and +16 relative to kb.
// ---------------------------------------------------------------------------
DEVINL v16bf load_afrag_bf16(const bf16* base, int ld, int row, int kb, int lane) {
  const bf16* p = base + (size_t)row * ld + kb + ((lane >> 4) << 3);
  AFrag f;
  f.q[0] = *(const uint4*)(p);
  f.q[1] = *(const uint4*)(p + 16);
  return f.v;
}

DEVINL v16bf load_bfrag_bf16(const bf16* base, int ld, int col, int kb, int lane) {
  const bf16* p = base + (size_t)col * ld + kb + ((lane >> 4) << 3);
  AFrag f;
  f.q[0] = *(const uint4*)(p);
  f.q[1] = *(const uint4*)(p + 16);
  return f.v;
}

DEVINL v16bf cvt16(const float* p0, const float* p1) {
  float t[16];
  *(float4*)(t + 0)  = *(const float4*)(p0);
  *(float4*)(t + 4)  = *(const float4*)(p0 + 4);
  *(float4*)(t + 8)  = *(const float4*)(p1);
  *(float4*)(t + 12) = *(const float4*)(p1 + 4);
  v16bf r;
#pragma unroll
  for (int i = 0; i < 16; ++i) r[i] = (bf16)t[i];
  return r;
}

DEVINL v16bf load_afrag_f32(const float* base, int ld, int row, int kb, int lane) {
  const float* p = base + (size_t)row * ld + kb + ((lane >> 4) << 3);
  return cvt16(p, p + 16);
}

DEVINL v16bf load_bfrag_f32(const float* base, int ld, int col, int kb, int lane) {
  const float* p = base + (size_t)col * ld + kb + ((lane >> 4) << 3);
  return cvt16(p, p + 16);
}

// ---------------------------------------------------------------------------
// Prefix-tuning MLP: for each sample, run ONLY the selected language's MLP
// (reference one-hot selects exactly one) and write kv position 0.
// ---------------------------------------------------------------------------
__global__ __launch_bounds__(256)
void prefix_kernel(const float* __restrict__ wte,
                   const float* __restrict__ w1, const float* __restrict__ b1,
                   const float* __restrict__ w2, const float* __restrict__ b2,
                   const int* __restrict__ lang,
                   bf16* __restrict__ kh, bf16* __restrict__ vT)
{
  __shared__ float emb_s[D_];
  __shared__ float h_s[BN];
  const int b = blockIdx.x;
  const int c = lang[b];

  const float* emb = wte + (size_t)c * D_;
  for (int i = threadIdx.x; i < D_; i += 256) emb_s[i] = emb[i];
  __syncthreads();

  const float* w1c = w1 + (size_t)c * BN * D_;
  for (int j = threadIdx.x; j < BN; j += 256) {
    const float* row = w1c + (size_t)j * D_;
    float acc = b1[c * BN + j];
    for (int d = 0; d < D_; d += 4) {
      float4 r4 = *(const float4*)(row + d);
      acc += r4.x * emb_s[d] + r4.y * emb_s[d + 1] +
             r4.z * emb_s[d + 2] + r4.w * emb_s[d + 3];
    }
    h_s[j] = tanhf(acc);
  }
  __syncthreads();

  const float* w2c = w2 + (size_t)c * (2 * D_) * BN;
  for (int e = threadIdx.x; e < 2 * D_; e += 256) {
    const float* row = w2c + (size_t)e * BN;
    float acc = b2[c * 2 * D_ + e];
    for (int d = 0; d < BN; d += 4) {
      float4 r4 = *(const float4*)(row + d);
      acc += r4.x * h_s[d] + r4.y * h_s[d + 1] +
             r4.z * h_s[d + 2] + r4.w * h_s[d + 3];
    }
    const bf16 val = (bf16)acc;
    if (e < D_) {                       // prefix key -> kh[b,h,0,hd]
      const int hh = e >> 6, hdi = e & 63;
      kh[((size_t)(b * H_ + hh) * LP + 0) * HD_ + hdi] = val;
    } else {                            // prefix value -> vT[b,h,hd,0]
      const int e2 = e - D_;
      const int hh = e2 >> 6, hdi = e2 & 63;
      vT[((size_t)(b * H_ + hh) * HD_ + hdi) * LP + 0] = val;
    }
  }
}

// ---------------------------------------------------------------------------
// Projection GEMM: Y[m,n] = sum_d A[m,d] * W[n,d] (+bias[n]) * scale
// MODE 0: -> qh  [B,H,T,hd]  bf16  (scale = 1/8)
// MODE 1: -> kh  [B,H,LP,hd] bf16  at t+1
// MODE 2: -> vT  [B,H,hd,LP] bf16  at t+1 (transposed for P*V B-fragments)
// MODE 3: -> out [M,D] fp32 (final output projection; A is bf16 ctx)
// Block tile 128x128, 8 waves of 32x64, K step 32, fused f32->bf16 convert.
// ---------------------------------------------------------------------------
template <int MODE>
__global__ __launch_bounds__(256)
void gemm_proj_kernel(const void* __restrict__ Ain,
                      const float* __restrict__ W,
                      const float* __restrict__ bias,
                      float scale, void* __restrict__ out)
{
  constexpr bool A_F32 = (MODE != 3);
  const int lane = threadIdx.x & 31;
  const int wid  = threadIdx.x >> 5;
  const int row0 = blockIdx.x * 128 + (wid >> 1) * 32;
  const int col0 = blockIdx.y * 128 + (wid & 1) * 64;
  const float* Af = (const float*)Ain;
  const bf16*  Ab = (const bf16*)Ain;
  const int ar = lane & 15;

  v8f acc[2][4];
#pragma unroll
  for (int i = 0; i < 2; ++i)
#pragma unroll
    for (int j = 0; j < 4; ++j) acc[i][j] = (v8f){0, 0, 0, 0, 0, 0, 0, 0};

  for (int kb = 0; kb < D_; kb += 32) {
    v16bf a0, a1;
    if (A_F32) {
      a0 = load_afrag_f32(Af, D_, row0 + ar, kb, lane);
      a1 = load_afrag_f32(Af, D_, row0 + 16 + ar, kb, lane);
    } else {
      a0 = load_afrag_bf16(Ab, D_, row0 + ar, kb, lane);
      a1 = load_afrag_bf16(Ab, D_, row0 + 16 + ar, kb, lane);
    }
    v16bf bw[4];
#pragma unroll
    for (int j = 0; j < 4; ++j)
      bw[j] = load_bfrag_f32(W, D_, col0 + j * 16 + ar, kb, lane);
#pragma unroll
    for (int j = 0; j < 4; ++j) {
      acc[0][j] = wmma_bf16(a0, bw[j], acc[0][j]);
      acc[1][j] = wmma_bf16(a1, bw[j], acc[1][j]);
    }
  }

  const int rbase = (lane >> 4) << 3;
#pragma unroll
  for (int i = 0; i < 2; ++i) {
#pragma unroll
    for (int j = 0; j < 4; ++j) {
      const int n = col0 + j * 16 + ar;
      const float bv = bias[n];
#pragma unroll
      for (int r = 0; r < 8; ++r) {
        const int m = row0 + i * 16 + rbase + r;
        const float val = (acc[i][j][r] + bv) * scale;
        if (MODE == 3) {
          ((float*)out)[(size_t)m * D_ + n] = val;   // [T,B,D] flat == [m,n]
        } else {
          const int bidx = m & 1;                    // m = t*B + b
          const int t    = m >> 1;
          const int hh   = n >> 6;
          const int hdi  = n & 63;
          bf16* o = (bf16*)out;
          if (MODE == 0)
            o[((size_t)(bidx * H_ + hh) * T_ + t) * HD_ + hdi] = (bf16)val;
          else if (MODE == 1)
            o[((size_t)(bidx * H_ + hh) * LP + (t + 1)) * HD_ + hdi] = (bf16)val;
          else
            o[((size_t)(bidx * H_ + hh) * HD_ + hdi) * LP + (t + 1)] = (bf16)val;
        }
      }
    }
  }
}

// ---------------------------------------------------------------------------
// Flash attention: one block per (qtile, head, batch); 8 waves, each wave owns
// 16 query rows and streams all LP keys in tiles of 32 with online softmax.
// Single running max shared per half-wave tile (softmax-equivalent), per-lane
// partial row sums reduced once after the key loop, masking only in the final
// tail tile, permlane16 butterflies for the cross-lane reductions.
// ---------------------------------------------------------------------------
__global__ __launch_bounds__(256)
void attn_kernel(const bf16* __restrict__ qh, const bf16* __restrict__ kh,
                 const bf16* __restrict__ vT, bf16* __restrict__ ctx)
{
  __shared__ __align__(16) bf16 pbuf[8][16 * 32];   // per-wave P staging tile
  const int lane = threadIdx.x & 31;
  const int wid  = threadIdx.x >> 5;
  const int h = blockIdx.y;
  const int b = blockIdx.z;
  const int qr0 = blockIdx.x * 128 + wid * 16;

  const bf16* Q = qh + (size_t)(b * H_ + h) * T_ * HD_;
  const bf16* K = kh + (size_t)(b * H_ + h) * LP * HD_;
  const bf16* V = vT + (size_t)(b * H_ + h) * HD_ * LP;

  const int ar = lane & 15;
  const int rbase = (lane >> 4) << 3;

  const v16bf aq0 = load_afrag_bf16(Q, HD_, qr0 + ar, 0, lane);
  const v16bf aq1 = load_afrag_bf16(Q, HD_, qr0 + ar, 32, lane);

  v8f acc[4];
#pragma unroll
  for (int n = 0; n < 4; ++n) acc[n] = (v8f){0, 0, 0, 0, 0, 0, 0, 0};
  float plocal[8];                    // per-lane partial row sums of exp()
#pragma unroll
  for (int r = 0; r < 8; ++r) plocal[r] = 0.0f;
  float mrun = -3.0e38f;              // running tile max (shared per half-wave)

  auto process_tile = [&](int j0, bool masked) {
    v8f s0 = (v8f){0, 0, 0, 0, 0, 0, 0, 0};
    v8f s1 = (v8f){0, 0, 0, 0, 0, 0, 0, 0};
    // scores: Q(16x64) x K^T(64x32) via two K=32 WMMAs per 16-key subtile
    const v16bf k00 = load_bfrag_bf16(K, HD_, j0 + ar, 0, lane);
    const v16bf k01 = load_bfrag_bf16(K, HD_, j0 + ar, 32, lane);
    const v16bf k10 = load_bfrag_bf16(K, HD_, j0 + 16 + ar, 0, lane);
    const v16bf k11 = load_bfrag_bf16(K, HD_, j0 + 16 + ar, 32, lane);
    s0 = wmma_bf16(aq0, k00, s0);
    s0 = wmma_bf16(aq1, k01, s0);
    s1 = wmma_bf16(aq0, k10, s1);
    s1 = wmma_bf16(aq1, k11, s1);

    if (masked) {
      const bool msk0 = (j0 + ar) >= LKV;       // lane owns cols j0+ar, j0+16+ar
      const bool msk1 = (j0 + 16 + ar) >= LKV;
#pragma unroll
      for (int r = 0; r < 8; ++r) {
        s0[r] = msk0 ? -1.0e30f : s0[r];
        s1[r] = msk1 ? -1.0e30f : s1[r];
      }
    }

    // one max over the whole 8x32 half-wave tile (safe: >= every row max)
    float mt = fmaxf(s0[0], s1[0]);
#pragma unroll
    for (int r = 1; r < 8; ++r) mt = fmaxf(mt, fmaxf(s0[r], s1[r]));
    mt = half_reduce_max(mt);

    const float mn   = fmaxf(mrun, mt);
    const float corr = __expf(mrun - mn);
    mrun = mn;
#pragma unroll
    for (int n = 0; n < 4; ++n) acc[n] *= corr;
#pragma unroll
    for (int r = 0; r < 8; ++r) {
      const float p0 = __expf(s0[r] - mn);
      const float p1 = __expf(s1[r] - mn);
      plocal[r] = plocal[r] * corr + (p0 + p1);
      // stage P (C-layout -> row-major LDS) for A-fragment reload
      pbuf[wid][(rbase + r) * 32 + ar]      = (bf16)p0;
      pbuf[wid][(rbase + r) * 32 + 16 + ar] = (bf16)p1;
    }
    asm volatile("s_wait_dscnt 0" ::: "memory");
    const v16bf pf = load_afrag_bf16(&pbuf[wid][0], 32, ar, 0, lane);
#pragma unroll
    for (int n = 0; n < 4; ++n) {
      const v16bf vf = load_bfrag_bf16(V, LP, n * 16 + ar, j0, lane);
      acc[n] = wmma_bf16(pf, vf, acc[n]);       // ctx += P(16x32) x V(32x16)
    }
  };

  for (int j0 = 0; j0 < LP - 32; j0 += 32) process_tile(j0, false);
  process_tile(LP - 32, true);                  // tail tile: keys 2048..2079

  // final per-row sum reduction (once, outside the key loop)
  float lrow[8];
#pragma unroll
  for (int r = 0; r < 8; ++r) lrow[r] = half_reduce_sum(plocal[r]);

#pragma unroll
  for (int n = 0; n < 4; ++n) {
    const int col = h * HD_ + n * 16 + ar;
#pragma unroll
    for (int r = 0; r < 8; ++r) {
      const int t = qr0 + rbase + r;
      ctx[((size_t)t * B_ + b) * D_ + col] = (bf16)(acc[n][r] / lrow[r]);
    }
  }
}

// ---------------------------------------------------------------------------
extern "C" void kernel_launch(void* const* d_in, const int* in_sizes, int n_in,
                              void* d_out, int out_size, void* d_ws, size_t ws_size,
                              hipStream_t stream)
{
  const float* query = (const float*)d_in[0];
  const float* key_  = (const float*)d_in[1];
  const float* value = (const float*)d_in[2];
  const float* wq = (const float*)d_in[3];
  const float* bq = (const float*)d_in[4];
  const float* wk = (const float*)d_in[5];
  const float* bk = (const float*)d_in[6];
  const float* wv = (const float*)d_in[7];
  const float* bv = (const float*)d_in[8];
  const float* wo = (const float*)d_in[9];
  const float* bo = (const float*)d_in[10];
  const float* wte   = (const float*)d_in[11];
  const float* ct_w1 = (const float*)d_in[12];
  const float* ct_b1 = (const float*)d_in[13];
  const float* ct_w2 = (const float*)d_in[14];
  const float* ct_b2 = (const float*)d_in[15];
  const int*   lang  = (const int*)d_in[16];

  const size_t QH_BYTES = (size_t)B_ * H_ * T_ * HD_ * sizeof(bf16);   // 8 MB
  const size_t KH_BYTES = (size_t)B_ * H_ * LP * HD_ * sizeof(bf16);   // 8.5 MB
  char* ws = (char*)d_ws;
  bf16* qh  = (bf16*)(ws);
  bf16* kh  = (bf16*)(ws + QH_BYTES);
  bf16* vT  = (bf16*)(ws + QH_BYTES + KH_BYTES);
  bf16* ctx = (bf16*)(ws + QH_BYTES + 2 * KH_BYTES);

  // zero kv buffers (prefix slot 0 written by prefix_kernel; tail stays 0 and
  // is masked to -inf in the attention kernel)
  hipMemsetAsync(kh, 0, 2 * KH_BYTES, stream);

  prefix_kernel<<<dim3(B_), dim3(256), 0, stream>>>(
      wte, ct_w1, ct_b1, ct_w2, ct_b2, lang, kh, vT);

  const dim3 gg(M_ / 128, D_ / 128);
  gemm_proj_kernel<0><<<gg, 256, 0, stream>>>(query, wq, bq, 0.125f, qh);
  gemm_proj_kernel<1><<<gg, 256, 0, stream>>>(key_,  wk, bk, 1.0f,  kh);
  gemm_proj_kernel<2><<<gg, 256, 0, stream>>>(value, wv, bv, 1.0f,  vT);

  attn_kernel<<<dim3(T_ / 128, H_, B_), 256, 0, stream>>>(qh, kh, vT, ctx);

  gemm_proj_kernel<3><<<gg, 256, 0, stream>>>(ctx, wo, bo, 1.0f, d_out);
}
